// RSSM_1692217115208
// MI455X (gfx1250) — matrix operand order repaired and checked by reading, not compiled, not run on hardware
//
#include <hip/hip_runtime.h>

// RSSM persistent-scan kernel for gfx1250 (MI455X).
// One wave32 owns a 16-row batch tile and runs the whole T=256 recurrence.
// GEMMs via v_wmma_f32_16x16x32_f16 (f16 in, f32 accum). All weight B-fragments
// are hoisted into registers for the whole scan (gfx1250 waves address up to
// 1024 VGPRs; total wave count is tiny so occupancy is not register-limited).
// Softmax/argmax are done in a transposed per-lane layout via a small LDS
// round-trip -> no cross-lane shuffles on the serial chain. Outputs are stored
// with non-temporal hints (write-once streams).

#define T_STEPS 256
#define NBATCH  4096
#define H       32
#define ZDIM    64      // Z_SIZE*Z_CLS
#define ADIM    16      // A_SIZE*A_CLS
#define EDIM    10      // X_SIZE*X_CLS
#define GOUT    96      // 3*H
#define ZA_K    96      // 80 padded to 3 K-tiles of 32
#define HE_K    64      // 42 padded to 2 K-tiles
#define DEC_K   96      // 96 = 3 K-tiles exactly
#define WAVES_PER_WG 2

typedef __attribute__((ext_vector_type(16))) _Float16 v16h;
typedef __attribute__((ext_vector_type(8)))  float    v8f;
typedef __attribute__((ext_vector_type(4)))  float    v4f;

__device__ __forceinline__ float sigm_f(float x) { return 1.0f / (1.0f + __expf(-x)); }
__device__ __forceinline__ float tanh_f(float x) { return 2.0f * sigm_f(2.0f * x) - 1.0f; }

__device__ __forceinline__ v8f wmma_f16(v16h a, v16h b, v8f c) {
    // (neg_a, A, neg_b, B, c_mod, C, reuse_a, reuse_b)
    return __builtin_amdgcn_wmma_f32_16x16x32_f16(false, a, false, b, (short)0, c,
                                                  false, false);
}

// A fragment (16x32 f16, MxK): lane half h: VGPR0-3 hold K = kt*32 + h*8 + {0..7},
// VGPR4-7 hold K = kt*32 + 16 + h*8 + {0..7}; M = lane&15. Buffer layout [M][Kd].
__device__ __forceinline__ v16h load_frag_A(const _Float16* buf, int Kd, int kt, int lane) {
    int m = lane & 15;
    int h = lane >> 4;
    union { v16h v; uint32_t u[8]; } r;
    const uint32_t* p = (const uint32_t*)(buf + m * Kd);   // Kd even -> 4B aligned
    int k0 = (kt * 32 + h * 8) >> 1;
    int k1 = k0 + 8;
#pragma unroll
    for (int i = 0; i < 4; ++i) {
        r.u[i]     = p[k0 + i];
        r.u[i + 4] = p[k1 + i];
    }
    return r.v;
}

// B fragment (32x16 f16, KxN): lane half h holds K = kt*32 + h*8 + {0..7}, kt*32 + 16 + h*8 + {0..7},
// column n = nt*16 + (lane&15). Weight buffer layout [n][Kd] (= torch W[out][in]).
__device__ __forceinline__ v16h load_frag_B(const _Float16* w, int Kd, int nt, int kt, int lane) {
    int n = nt * 16 + (lane & 15);
    int h = lane >> 4;
    union { v16h v; uint32_t u[8]; } r;
    const uint32_t* p = (const uint32_t*)(w + n * Kd);
    int kb = (kt * 32 + h * 8) >> 1;
#pragma unroll
    for (int i = 0; i < 8; ++i) r.u[i] = p[kb + (i & 3) + (i >> 2) * 8];
    return r.v;
}

__global__ __launch_bounds__(WAVES_PER_WG * 32)
void rssm_scan_kernel(const float* __restrict__ x,    const float* __restrict__ a,
                      const float* __restrict__ h0,   const float* __restrict__ gum,
                      const float* __restrict__ Wih,  const float* __restrict__ bih,
                      const float* __restrict__ Whh,  const float* __restrict__ bhh,
                      const float* __restrict__ Wenc, const float* __restrict__ benc,
                      const float* __restrict__ Wdec, const float* __restrict__ bdec,
                      float* __restrict__ out_x, float* __restrict__ out_h,
                      float* __restrict__ out_z)
{
    __shared__ _Float16 sWih [GOUT * ZA_K];                 // 18 KB
    __shared__ _Float16 sWhh [GOUT * H];                    //  6 KB
    __shared__ _Float16 sWenc[ZDIM * HE_K];                 //  8 KB
    __shared__ _Float16 sWdec[16   * DEC_K];                //  3 KB (10 rows padded)
    __shared__ _Float16 sZA [WAVES_PER_WG][16 * ZA_K];      // [z(64), a(16), pad(16)]
    __shared__ _Float16 sHE [WAVES_PER_WG][16 * HE_K];      // [h(32), e(10), pad(22)]
    __shared__ _Float16 sDEC[WAVES_PER_WG][16 * DEC_K];     // [h(32), z(64)]
    __shared__ float    sLOG[WAVES_PER_WG][16 * ZDIM];      // encoder logits scratch

    const int tid = threadIdx.x;

    // --- cooperative weight conversion f32 -> f16 into LDS (zero-padded) ---
    for (int i = tid; i < GOUT * ZA_K; i += blockDim.x) {
        int k = i % ZA_K;
        sWih[i] = (_Float16)((k < 80) ? Wih[(i / ZA_K) * 80 + k] : 0.0f);
    }
    for (int i = tid; i < GOUT * H; i += blockDim.x) sWhh[i] = (_Float16)Whh[i];
    for (int i = tid; i < ZDIM * HE_K; i += blockDim.x) {
        int k = i % HE_K;
        sWenc[i] = (_Float16)((k < 42) ? Wenc[(i / HE_K) * 42 + k] : 0.0f);
    }
    for (int i = tid; i < 16 * DEC_K; i += blockDim.x) {
        int n = i / DEC_K;
        sWdec[i] = (_Float16)((n < 10) ? Wdec[i] : 0.0f);
    }
    __syncthreads();

    const int wave = tid >> 5;
    const int lane = tid & 31;
    const int col  = lane & 15;
    const int hlf  = lane >> 4;
    const int m0   = (blockIdx.x * WAVES_PER_WG + wave) * 16;

    _Float16* ZA  = sZA[wave];
    _Float16* HE  = sHE[wave];
    _Float16* DEC = sDEC[wave];
    float*    LOG = sLOG[wave];

    // zero the K-padding once (stays zero for the whole scan)
    for (int i = lane; i < 16 * 16; i += 32)
        ZA[(i >> 4) * ZA_K + 80 + (i & 15)] = (_Float16)0.0f;
    for (int i = lane; i < 16 * 22; i += 32)
        HE[(i / 22) * HE_K + 42 + (i % 22)] = (_Float16)0.0f;

    // --- hoist ALL weight B-fragments into registers for the whole scan ---
    // 35 fragments x 8 VGPRs = 280 VGPRs; fine at this wave count (<=1024/wave).
    v16h wih_f[3][6], whh_f[6], wenc_f[2][4], wdec_f[3];
#pragma unroll
    for (int kt = 0; kt < 3; ++kt)
#pragma unroll
        for (int nt = 0; nt < 6; ++nt)
            wih_f[kt][nt] = load_frag_B(sWih, ZA_K, nt, kt, lane);
#pragma unroll
    for (int nt = 0; nt < 6; ++nt) whh_f[nt] = load_frag_B(sWhh, H, nt, 0, lane);
#pragma unroll
    for (int kt = 0; kt < 2; ++kt)
#pragma unroll
        for (int nt = 0; nt < 4; ++nt)
            wenc_f[kt][nt] = load_frag_B(sWenc, HE_K, nt, kt, lane);
#pragma unroll
    for (int kt = 0; kt < 3; ++kt) wdec_f[kt] = load_frag_B(sWdec, DEC_K, 0, kt, lane);

    // per-lane bias registers (bias depends on output column only)
    float bgi_r[6], bgh_r[6], ben_r[4], bde_r;
#pragma unroll
    for (int nt = 0; nt < 6; ++nt) { bgi_r[nt] = bih[nt * 16 + col]; bgh_r[nt] = bhh[nt * 16 + col]; }
#pragma unroll
    for (int nt = 0; nt < 4; ++nt) ben_r[nt] = benc[nt * 16 + col];
    bde_r = (col < 10) ? bdec[col] : 0.0f;

    // --- h state in accumulator layout: hf[jt][e] -> (M = e + 8*hlf, n = jt*16+col) ---
    v8f hf[2];
#pragma unroll
    for (int jt = 0; jt < 2; ++jt) {
#pragma unroll
        for (int e = 0; e < 8; ++e) {
            int m  = m0 + e + 8 * hlf;
            float v = h0[m * H + jt * 16 + col];
            hf[jt][e] = v;
            __builtin_nontemporal_store(v, &out_h[(size_t)m * H + jt * 16 + col]); // t=0
            _Float16 hv = (_Float16)v;
            HE [(e + 8 * hlf) * HE_K  + jt * 16 + col] = hv;
            DEC[(e + 8 * hlf) * DEC_K + jt * 16 + col] = hv;
        }
    }

    auto load_e = [&](int t) {  // x[t] flattened -> HE cols 32..41
        for (int i = lane; i < 16 * EDIM; i += 32) {
            int r = i / EDIM, c = i % EDIM;
            HE[r * HE_K + H + c] = (_Float16)x[((size_t)t * NBATCH + m0 + r) * EDIM + c];
        }
    };

    // logits = [h,e] @ Wenc^T -> LDS scratch -> transposed per-lane
    // gumbel-argmax + softmax (pure VALU, no cross-lane ops).
    auto encode = [&](int t) {
        v8f lf[4];
#pragma unroll
        for (int nt = 0; nt < 4; ++nt) { v8f c; for (int e = 0; e < 8; ++e) c[e] = ben_r[nt]; lf[nt] = c; }
#pragma unroll
        for (int kt = 0; kt < 2; ++kt) {
            v16h A = load_frag_A(HE, HE_K, kt, lane);
#pragma unroll
            for (int nt = 0; nt < 4; ++nt)
                lf[nt] = wmma_f16(A, wenc_f[kt][nt], lf[nt]);
        }
        // spill fragment-layout logits to LDS scratch
#pragma unroll
        for (int nt = 0; nt < 4; ++nt)
#pragma unroll
            for (int e = 0; e < 8; ++e)
                LOG[(e + 8 * hlf) * ZDIM + nt * 16 + col] = lf[nt][e];

        // transposed pass: each lane owns 4 (row, group) pairs; 8 classes serial
#pragma unroll
        for (int p = 0; p < 4; ++p) {
            int m  = p * 4 + (lane >> 3);   // 0..15
            int g  = lane & 7;              // group 0..7
            int mg = m0 + m;
            v4f la = *reinterpret_cast<const v4f*>(&LOG[m * ZDIM + g * 8]);
            v4f lb = *reinterpret_cast<const v4f*>(&LOG[m * ZDIM + g * 8 + 4]);
            const float* gp = &gum[((size_t)t * NBATCH + mg) * ZDIM + g * 8];
            v4f ga = *reinterpret_cast<const v4f*>(gp);
            v4f gb = *reinterpret_cast<const v4f*>(gp + 4);
            float lg[8] = {la[0], la[1], la[2], la[3], lb[0], lb[1], lb[2], lb[3]};
            float gu[8] = {ga[0], ga[1], ga[2], ga[3], gb[0], gb[1], gb[2], gb[3]};
            // max for softmax stability
            float mx = lg[0];
#pragma unroll
            for (int c = 1; c < 8; ++c) mx = fmaxf(mx, lg[c]);
            // gumbel-argmax (strict > keeps lowest index on ties, like jnp.argmax)
            float bk = lg[0] + gu[0]; int bc = 0;
#pragma unroll
            for (int c = 1; c < 8; ++c) {
                float k = lg[c] + gu[c];
                if (k > bk) { bk = k; bc = c; }
            }
            float ex[8], sum = 0.0f;
#pragma unroll
            for (int c = 0; c < 8; ++c) { ex[c] = __expf(lg[c] - mx); sum += ex[c]; }
            float rs = 1.0f / sum;          // one divide per group, then multiplies
            float zv[8];
            union { _Float16 h[8]; uint4 u; } pk;
#pragma unroll
            for (int c = 0; c < 8; ++c) {
                float prob = ex[c] * rs;
                float onehot = (c == bc) ? 1.0f : 0.0f;
                zv[c] = (onehot + prob) - prob;   // straight-through forward value
                pk.h[c] = (_Float16)zv[c];
            }
            float* zo = &out_z[((size_t)t * NBATCH + mg) * ZDIM + g * 8];
            v4f z0v = {zv[0], zv[1], zv[2], zv[3]};
            v4f z1v = {zv[4], zv[5], zv[6], zv[7]};
            __builtin_nontemporal_store(z0v, reinterpret_cast<v4f*>(zo));
            __builtin_nontemporal_store(z1v, reinterpret_cast<v4f*>(zo + 4));
            *reinterpret_cast<uint4*>(&ZA [m * ZA_K  + g * 8])     = pk.u;  // GRU input
            *reinterpret_cast<uint4*>(&DEC[m * DEC_K + H + g * 8]) = pk.u;  // decoder input
        }
    };

    auto decode = [&](int t) {  // x_logits = [h,z] @ Wdec^T + b
        v8f c; for (int e = 0; e < 8; ++e) c[e] = bde_r;
#pragma unroll
        for (int kt = 0; kt < 3; ++kt)
            c = wmma_f16(load_frag_A(DEC, DEC_K, kt, lane), wdec_f[kt], c);
        if (col < 10) {
#pragma unroll
            for (int e = 0; e < 8; ++e) {
                int m = m0 + e + 8 * hlf;
                __builtin_nontemporal_store(c[e], &out_x[((size_t)t * NBATCH + m) * EDIM + col]);
            }
        }
    };

    // ---- t = 0 ----
    load_e(0);
    encode(0);
    decode(0);

    // ---- recurrence t = 1..T-1 ----
    for (int t = 1; t < T_STEPS; ++t) {
        // pull next step's operands toward the caches under the serial chain
        if (t + 1 < T_STEPS) {
            __builtin_prefetch(&gum[((size_t)(t + 1) * NBATCH + m0) * ZDIM + lane * 32], 0, 0);
            __builtin_prefetch(&x  [((size_t)(t + 1) * NBATCH + m0) * EDIM + lane * 5],  0, 0);
            __builtin_prefetch(&a  [((size_t)t       * NBATCH + m0) * ADIM + lane * 8],  0, 0);
        }
        // a[t-1] -> ZA cols 64..79
        for (int i = lane; i < 16 * ADIM; i += 32) {
            int r = i >> 4, c2 = i & 15;
            ZA[r * ZA_K + ZDIM + c2] =
                (_Float16)a[((size_t)(t - 1) * NBATCH + m0 + r) * ADIM + c2];
        }
        load_e(t);

        // gi = za @ Wih^T + b_ih   (3 K-tiles x 6 N-tiles)
        v8f gi[6], gh[6];
#pragma unroll
        for (int nt = 0; nt < 6; ++nt) { v8f c; for (int e = 0; e < 8; ++e) c[e] = bgi_r[nt]; gi[nt] = c; }
#pragma unroll
        for (int kt = 0; kt < 3; ++kt) {
            v16h A = load_frag_A(ZA, ZA_K, kt, lane);
#pragma unroll
            for (int nt = 0; nt < 6; ++nt)
                gi[nt] = wmma_f16(A, wih_f[kt][nt], gi[nt]);
        }
        // gh = h @ Whh^T + b_hh    (reads h_{t-1} from HE cols 0..31)
        {
            v16h A = load_frag_A(HE, HE_K, 0, lane);
#pragma unroll
            for (int nt = 0; nt < 6; ++nt) {
                v8f c; for (int e = 0; e < 8; ++e) c[e] = bgh_r[nt];
                gh[nt] = wmma_f16(A, whh_f[nt], c);
            }
        }
        // GRU gates -> h_t ; write back (registers + LDS f16 + global f32)
#pragma unroll
        for (int jt = 0; jt < 2; ++jt) {
            v8f hn;
#pragma unroll
            for (int e = 0; e < 8; ++e) {
                float r = sigm_f(gi[jt][e]     + gh[jt][e]);
                float z = sigm_f(gi[jt + 2][e] + gh[jt + 2][e]);
                float n = tanh_f(gi[jt + 4][e] + r * gh[jt + 4][e]);
                hn[e] = (1.0f - z) * n + z * hf[jt][e];
            }
            hf[jt] = hn;
#pragma unroll
            for (int e = 0; e < 8; ++e) {
                int m = m0 + e + 8 * hlf;
                float hv = hn[e];
                __builtin_nontemporal_store(hv, &out_h[(size_t)t * NBATCH * H + m * H + jt * 16 + col]);
                _Float16 hh = (_Float16)hv;
                HE [(e + 8 * hlf) * HE_K  + jt * 16 + col] = hh;
                DEC[(e + 8 * hlf) * DEC_K + jt * 16 + col] = hh;
            }
        }

        encode(t);
        decode(t);
    }
}

extern "C" void kernel_launch(void* const* d_in, const int* in_sizes, int n_in,
                              void* d_out, int out_size, void* d_ws, size_t ws_size,
                              hipStream_t stream) {
    const float* x    = (const float*)d_in[0];
    const float* a    = (const float*)d_in[1];
    const float* h0   = (const float*)d_in[2];
    const float* gum  = (const float*)d_in[3];
    const float* Wih  = (const float*)d_in[4];
    const float* bih  = (const float*)d_in[5];
    const float* Whh  = (const float*)d_in[6];
    const float* bhh  = (const float*)d_in[7];
    const float* Wenc = (const float*)d_in[8];
    const float* benc = (const float*)d_in[9];
    const float* Wdec = (const float*)d_in[10];
    const float* bdec = (const float*)d_in[11];

    float* out_x = (float*)d_out;                               // [T,N,5,2]
    float* out_h = out_x + (size_t)T_STEPS * NBATCH * EDIM;     // [T,N,32]
    float* out_z = out_h + (size_t)T_STEPS * NBATCH * H;        // [T,N,8,8]

    dim3 grid(NBATCH / (16 * WAVES_PER_WG));   // 128 workgroups
    dim3 block(32 * WAVES_PER_WG);             // 2 waves / WG (~60 KB LDS)
    hipLaunchKernelGGL(rssm_scan_kernel, grid, block, 0, stream,
                       x, a, h0, gum, Wih, bih, Whh, bhh,
                       Wenc, benc, Wdec, bdec, out_x, out_h, out_z);
}
